// BayesianRNN_46875273068619
// MI455X (gfx1250) — matrix operand order repaired and checked by reading, not compiled, not run on hardware
//
#include <hip/hip_runtime.h>
#include <hip/hip_bf16.h>
#include <stdint.h>
#include <stddef.h>

// ---------------- problem constants ----------------
#define BB     256          // batch
#define TT     128          // timesteps
#define LL     16           // input dim
#define UU     512          // hidden
#define NN     (4*UU)       // 2048 gate width
#define KREAL  (UU+LL)      // 528
#define KPAD   544          // 17 * 32
#define KC     32           // WMMA K per step
#define KCHUNKS (KPAD/KC)   // 17
#define TILE_M 64
#define TILE_N 128
#define NWELEM (KREAL*NN)            // 1081344 = 4224*256
#define SAMPLE_BLOCKS (NWELEM/256)   // 4224
#define HS_ELEMS (BB*TT*UU)          // 16777216

typedef __attribute__((ext_vector_type(16))) __bf16 v16bf;
typedef __attribute__((ext_vector_type(8)))  float  v8f;

// ---- gfx1250 async global->LDS path (probe via __has_builtin) ----
#if defined(__has_builtin)
# if __has_builtin(__builtin_amdgcn_global_load_async_to_lds_b128)
#  define USE_ASYNC_LDS 1
# endif
#endif
#ifndef USE_ASYNC_LDS
# define USE_ASYNC_LDS 0
#endif

#if USE_ASYNC_LDS
// exact parameter types per hipcc diagnostic:
//   arg0: __attribute__((vector_size(16))) int  addrspace(1)*   (global src)
//   arg1: same vector type, addrspace(3)                        (LDS dst)
typedef __attribute__((vector_size(16))) int async_v4i;
typedef __attribute__((address_space(1))) async_v4i* g1_v4i;
typedef __attribute__((address_space(3))) async_v4i* l3_v4i;
#endif

// ---------------- threefry2x32 (host+device) ----------------
__host__ __device__ __forceinline__ uint32_t rotl32(uint32_t x, uint32_t r) {
    return (x << r) | (x >> (32u - r));
}

__host__ __device__ inline uint2 threefry2x32(uint2 key, uint2 ctr) {
    uint32_t k0 = key.x, k1 = key.y;
    uint32_t k2 = k0 ^ k1 ^ 0x1BD11BDAu;
    uint32_t x0 = ctr.x + k0, x1 = ctr.y + k1;
    const uint32_t RA[4] = {13u, 15u, 26u, 6u};
    const uint32_t RB[4] = {17u, 29u, 16u, 24u};
    #pragma unroll
    for (int i = 0; i < 4; ++i) { x0 += x1; x1 = rotl32(x1, RA[i]); x1 ^= x0; }
    x0 += k1; x1 += k2 + 1u;
    #pragma unroll
    for (int i = 0; i < 4; ++i) { x0 += x1; x1 = rotl32(x1, RB[i]); x1 ^= x0; }
    x0 += k2; x1 += k0 + 2u;
    #pragma unroll
    for (int i = 0; i < 4; ++i) { x0 += x1; x1 = rotl32(x1, RA[i]); x1 ^= x0; }
    x0 += k0; x1 += k1 + 3u;
    #pragma unroll
    for (int i = 0; i < 4; ++i) { x0 += x1; x1 = rotl32(x1, RB[i]); x1 ^= x0; }
    x0 += k1; x1 += k2 + 4u;
    #pragma unroll
    for (int i = 0; i < 4; ++i) { x0 += x1; x1 = rotl32(x1, RA[i]); x1 ^= x0; }
    x0 += k2; x1 += k0 + 5u;
    return make_uint2(x0, x1);
}

// ---------------- device math helpers ----------------
#define NEG_HALF_LOG2PI (-0.9189385332046727f)
#define LOG_09          (-0.10536051565782628f)   // log1p(-0.1)
#define LOG_01          (-2.302585092994046f)     // log(0.1)
#define INV_2S2         (81377.39570950196f)      // exp(12)/2  (sigma2 = e^-6)

__device__ __forceinline__ float softplusf(float x) {
    return (x > 20.0f) ? x : log1pf(expf(x));
}
__device__ __forceinline__ float sigmoidf(float x) {
    return 1.0f / (1.0f + expf(-x));
}
__device__ __forceinline__ float logaddexpf(float a, float b) {
    float m = fmaxf(a, b);
    float d = fminf(a, b) - m;
    return m + log1pf(expf(d));
}
__device__ __forceinline__ float mix_logprob(float x) {
    float x2 = x * x;
    float lp1 = LOG_09 + NEG_HALF_LOG2PI - 0.5f * x2;            // sigma1 = 1
    float lp2 = LOG_01 + NEG_HALF_LOG2PI + 6.0f - x2 * INV_2S2;  // sigma2 = e^-6
    return logaddexpf(lp1, lp2);
}
__device__ __forceinline__ float normal_from(uint2 r) {
    float u1 = (float)(r.x >> 8) * 0x1p-24f + 0x1p-25f;  // (0,1)
    float u2 = (float)(r.y >> 8) * 0x1p-24f;
    float rad = sqrtf(-2.0f * logf(u1));
    return rad * cosf(6.2831853071795864f * u2);
}

// ======================================================================
// init: zero c, Abuf (=> h0 = 0 and K-pad columns), W K-pad rows, accum
// ======================================================================
__global__ __launch_bounds__(256)
void init_kernel(float* __restrict__ cbuf, __bf16* __restrict__ Abuf,
                 __bf16* __restrict__ Wb, float* __restrict__ accum) {
    int i = blockIdx.x * 256 + threadIdx.x;          // 544 blocks -> 139264
    if (i < BB * UU)   cbuf[i] = 0.0f;
    if (i < BB * KPAD) Abuf[i] = (__bf16)0.0f;
    if (i < NN * (KPAD - KREAL)) {                   // 2048 * 16 pad rows
        int n = i >> 4, k = KREAL + (i & 15);
        Wb[n * KPAD + k] = (__bf16)0.0f;
    }
    if (i < 2) accum[i] = 0.0f;
}

// ======================================================================
// sample: W_t (bf16, N-major, K-padded) + Bw_t, accumulate lp/lq partials
// ======================================================================
__global__ __launch_bounds__(256)
void sample_kernel(const float* __restrict__ Wmu, const float* __restrict__ Wrho,
                   const float* __restrict__ Bmu, const float* __restrict__ Brho,
                   __bf16* __restrict__ Wb, float* __restrict__ Bwbuf,
                   float* __restrict__ partials, uint2 kW, uint2 kB) {
    int gid = blockIdx.x * 256 + threadIdx.x;        // exactly NWELEM threads
    int k = gid >> 11;                               // row in [0,528)
    int n = gid & 2047;                              // col in [0,2048)

    float mu  = Wmu[gid];
    float rho = Wrho[gid];
    float sig = softplusf(rho);
    float eps = normal_from(threefry2x32(kW, make_uint2((uint32_t)gid, 0u)));
    float w   = mu + sig * eps;
    Wb[n * KPAD + k] = (__bf16)w;                    // transposed (N-major) for WMMA B

    float lp = mix_logprob(w);
    float lq = NEG_HALF_LOG2PI - logf(sig) - 0.5f * eps * eps;

    if (gid < NN) {                                  // fold bias sampling into first 2048 lanes
        float bsig = softplusf(Brho[gid]);
        float beps = normal_from(threefry2x32(kB, make_uint2((uint32_t)gid, 1u)));
        float bw   = Bmu[gid] + bsig * beps;
        Bwbuf[gid] = bw;
        lp += mix_logprob(bw);
        lq += NEG_HALF_LOG2PI - logf(bsig) - 0.5f * beps * beps;
    }

    __shared__ float sh0[256];
    __shared__ float sh1[256];
    sh0[threadIdx.x] = lp; sh1[threadIdx.x] = lq;
    __syncthreads();
    #pragma unroll
    for (int s = 128; s > 0; s >>= 1) {
        if (threadIdx.x < s) {
            sh0[threadIdx.x] += sh0[threadIdx.x + s];
            sh1[threadIdx.x] += sh1[threadIdx.x + s];
        }
        __syncthreads();
    }
    if (threadIdx.x == 0) {
        partials[2 * blockIdx.x]     = sh0[0];
        partials[2 * blockIdx.x + 1] = sh1[0];
    }
}

// deterministic single-block reduction of partials into running accumulators
__global__ __launch_bounds__(256)
void reduce_kernel(const float* __restrict__ partials, float* __restrict__ accum) {
    __shared__ float sh0[256];
    __shared__ float sh1[256];
    float lp = 0.0f, lq = 0.0f;
    for (int i = threadIdx.x; i < SAMPLE_BLOCKS; i += 256) {
        lp += partials[2 * i];
        lq += partials[2 * i + 1];
    }
    sh0[threadIdx.x] = lp; sh1[threadIdx.x] = lq;
    __syncthreads();
    #pragma unroll
    for (int s = 128; s > 0; s >>= 1) {
        if (threadIdx.x < s) {
            sh0[threadIdx.x] += sh0[threadIdx.x + s];
            sh1[threadIdx.x] += sh1[threadIdx.x + s];
        }
        __syncthreads();
    }
    if (threadIdx.x == 0) { accum[0] += sh0[0]; accum[1] += sh1[0]; }
}

// ======================================================================
// pack x_t into bf16 A-buffer columns [0,16)
// ======================================================================
__global__ __launch_bounds__(256)
void pack_x_kernel(const float* __restrict__ x, __bf16* __restrict__ Abuf, int t) {
    int i = blockIdx.x * 256 + threadIdx.x;          // 4096 threads
    if (i < BB * LL) {
        int b = i >> 4, l = i & 15;
        Abuf[b * KPAD + l] = (__bf16)x[(b * TT + t) * LL + l];
    }
}

// ======================================================================
// GEMM: z[256x2048] = A[256x544]bf16 @ W[544x2048]bf16 + Bw, via WMMA
// block tile 64x128, 8 waves, each wave 16x64 (4 accumulators).
// Double-buffered LDS; async global->LDS loads for chunk c+1 stay in
// flight (s_wait_asynccnt 3) while chunk c is fed to the matrix pipes.
// ======================================================================
__global__ __launch_bounds__(256)
void gemm_kernel(const __bf16* __restrict__ Abuf,   // [BB][KPAD] row-major
                 const __bf16* __restrict__ Wb,     // [NN][KPAD] N-major
                 const float*  __restrict__ Bw,     // [NN]
                 float* __restrict__ z) {            // [BB][NN]
    __shared__ __align__(16) __bf16 As[2][TILE_M * KC];
    __shared__ __align__(16) __bf16 Bs[2][TILE_N * KC];

    const int tid  = threadIdx.x;
    const int wave = tid >> 5;
    const int lane = tid & 31;
    const int bm   = blockIdx.x * TILE_M;
    const int bn   = blockIdx.y * TILE_N;
    const int m_off = (wave & 3) * 16;
    const int n_off = (wave >> 2) * 64;

    const v8f vzero = {0.f, 0.f, 0.f, 0.f, 0.f, 0.f, 0.f, 0.f};
    v8f acc[4];
    #pragma unroll
    for (int s = 0; s < 4; ++s) acc[s] = vzero;

    // staging coordinates: one 16B chunk per (tid, slot)
    const int a_mt = tid >> 2, a_k8 = (tid & 3) * 8;
    const int b_nt0 = tid >> 2,         b_k80 = (tid & 3) * 8;
    const int b_nt1 = (tid + 256) >> 2, b_k81 = b_k80;   // second 16B slot

    // ---- stage one K-chunk (3 async ops per wave) into buffer `buf` ----
    auto stage = [&](int kc, int buf) {
        {
            const __bf16* g = &Abuf[(bm + a_mt) * KPAD + kc + a_k8];
            __bf16*       l = &As[buf][a_mt * KC + a_k8];
#if USE_ASYNC_LDS
            __builtin_amdgcn_global_load_async_to_lds_b128((g1_v4i)g, (l3_v4i)l, 0, 0);
#else
            *(uint4*)l = *(const uint4*)g;
#endif
        }
        {
            const __bf16* g = &Wb[(bn + b_nt0) * KPAD + kc + b_k80];
            __bf16*       l = &Bs[buf][b_nt0 * KC + b_k80];
#if USE_ASYNC_LDS
            __builtin_amdgcn_global_load_async_to_lds_b128((g1_v4i)g, (l3_v4i)l, 0, 0);
#else
            *(uint4*)l = *(const uint4*)g;
#endif
        }
        {
            const __bf16* g = &Wb[(bn + b_nt1) * KPAD + kc + b_k81];
            __bf16*       l = &Bs[buf][b_nt1 * KC + b_k81];
#if USE_ASYNC_LDS
            __builtin_amdgcn_global_load_async_to_lds_b128((g1_v4i)g, (l3_v4i)l, 0, 0);
#else
            *(uint4*)l = *(const uint4*)g;
#endif
        }
        if (kc + KC < KPAD)  // cover L2->WGP latency for the chunk after next
            __builtin_prefetch(&Wb[(bn + b_nt0) * KPAD + kc + KC + b_k80], 0, 1);
    };

    stage(0, 0);                                     // prologue: fill buffer 0

    for (int c = 0; c < KCHUNKS; ++c) {
        const int cur = c & 1;
        if (c + 1 < KCHUNKS) {
            stage((c + 1) * KC, cur ^ 1);            // software pipeline: next chunk
#if USE_ASYNC_LDS
            __builtin_amdgcn_s_wait_asynccnt(3);     // older 3 (chunk c) landed; 3 in flight
#endif
        } else {
#if USE_ASYNC_LDS
            __builtin_amdgcn_s_wait_asynccnt(0);
#endif
        }
        __syncthreads();                             // LDS fill visible block-wide

        // ---- A fragment: 16x32 bf16, documented wave32 layout ----
        union { v16bf v; uint32_t u[8]; } af;
        {
            int m   = m_off + (lane & 15);
            int kbA = (lane >> 4) * 8;
            #pragma unroll
            for (int j = 0; j < 8; ++j) {
                int k = ((j < 4) ? (2 * j) : (16 + 2 * (j - 4))) + kbA;
                af.u[j] = *(const uint32_t*)&As[cur][m * KC + k];
            }
        }
        // ---- 4 x (B fragment + WMMA) ----
        #pragma unroll
        for (int s = 0; s < 4; ++s) {
            union { v16bf v; uint32_t u[8]; } bfr;
            int n   = n_off + s * 16 + (lane & 15);
            int kbB = (lane >> 4) * 8;
            #pragma unroll
            for (int j = 0; j < 8; ++j)
                bfr.u[j] = *(const uint32_t*)&Bs[cur][n * KC + kbB + ((j < 4) ? (2 * j) : (16 + 2 * (j - 4)))];
            acc[s] = __builtin_amdgcn_wmma_f32_16x16x32_bf16(
                false, af.v, false, bfr.v, (short)0, acc[s], false, false);
        }
        __syncthreads();                             // done reading `cur` before overwrite
    }

    // ---- epilogue: + bias, store fp32 z (C layout: VGPR r -> M=r / r+8) ----
    #pragma unroll
    for (int s = 0; s < 4; ++s) {
        int n_g  = bn + n_off + s * 16 + (lane & 15);
        float bias = Bw[n_g];
        int m_hi = (lane >> 4) * 8;
        #pragma unroll
        for (int r = 0; r < 8; ++r) {
            int m_g = bm + m_off + m_hi + r;
            z[m_g * NN + n_g] = acc[s][r] + bias;
        }
    }
}

// ======================================================================
// gates: LSTM cell update, write h to output + back into bf16 A-buffer
// ======================================================================
__global__ __launch_bounds__(256)
void gates_kernel(const float* __restrict__ z, float* __restrict__ c,
                  float* __restrict__ hs_out, __bf16* __restrict__ Abuf, int t) {
    int i = blockIdx.x * 256 + threadIdx.x;          // 131072 threads
    int b = i >> 9, u = i & 511;
    const float* zr = &z[b * NN];
    float zi = zr[u], zj = zr[UU + u], zf = zr[2 * UU + u], zo = zr[3 * UU + u];
    float cn = c[i] * sigmoidf(zf) + sigmoidf(zi) * tanhf(zj);
    float h  = tanhf(cn) * sigmoidf(zo);
    c[i] = cn;
    hs_out[(b * TT + t) * UU + u] = h;
    Abuf[b * KPAD + LL + u] = (__bf16)h;
}

__global__ void finalize_kernel(const float* __restrict__ accum, float* __restrict__ out_tail) {
    if (threadIdx.x == 0) { out_tail[0] = accum[0]; out_tail[1] = accum[1]; }
}

// ======================================================================
// host: orchestrate T sequential steps on `stream`
// ======================================================================
extern "C" void kernel_launch(void* const* d_in, const int* in_sizes, int n_in,
                              void* d_out, int out_size, void* d_ws, size_t ws_size,
                              hipStream_t stream) {
    const float* x     = (const float*)d_in[0];
    const float* W_mu  = (const float*)d_in[1];
    const float* W_rho = (const float*)d_in[2];
    const float* B_mu  = (const float*)d_in[3];
    const float* B_rho = (const float*)d_in[4];
    float* out = (float*)d_out;

    // workspace carve-up (256B-aligned)
    char* ws = (char*)d_ws;
    size_t off = 0;
    __bf16* Wb     = (__bf16*)(ws + off); off += (size_t)NN * KPAD * 2;      // 2,228,224
    __bf16* Abuf   = (__bf16*)(ws + off); off += (size_t)BB * KPAD * 2;      //   278,528
    float*  zbuf   = (float*) (ws + off); off += (size_t)BB * NN * 4;        // 2,097,152
    float*  cbuf   = (float*) (ws + off); off += (size_t)BB * UU * 4;        //   524,288
    float*  Bwbuf  = (float*) (ws + off); off += (size_t)NN * 4;             //     8,192
    float*  parts  = (float*) (ws + off); off += (size_t)SAMPLE_BLOCKS * 2 * 4;
    float*  accum  = (float*) (ws + off); off += 256;
    (void)ws_size; (void)in_sizes; (void)n_in; (void)out_size;

    init_kernel<<<(BB * KPAD + 255) / 256, 256, 0, stream>>>(cbuf, Abuf, Wb, accum);

    const uint2 base_key = make_uint2(42u, 0u);
    const dim3 gemm_grid(BB / TILE_M, NN / TILE_N);   // 4 x 16

    for (int t = 0; t < TT; ++t) {
        uint2 kW = threefry2x32(base_key, make_uint2((uint32_t)(2 * t), 0u));
        uint2 kB = threefry2x32(base_key, make_uint2((uint32_t)(2 * t + 1), 0u));

        sample_kernel<<<SAMPLE_BLOCKS, 256, 0, stream>>>(
            W_mu, W_rho, B_mu, B_rho, Wb, Bwbuf, parts, kW, kB);
        pack_x_kernel<<<(BB * LL + 255) / 256, 256, 0, stream>>>(x, Abuf, t);
        gemm_kernel<<<gemm_grid, 256, 0, stream>>>(Abuf, Wb, Bwbuf, zbuf);
        gates_kernel<<<(BB * UU) / 256, 256, 0, stream>>>(zbuf, cbuf, out, Abuf, t);
        reduce_kernel<<<1, 256, 0, stream>>>(parts, accum);
    }
    finalize_kernel<<<1, 64, 0, stream>>>(accum, out + HS_ELEMS);
}
